// CrossAttention_50208167690251
// MI455X (gfx1250) — matrix operand order repaired and checked by reading, not compiled, not run on hardware
//
#include <hip/hip_runtime.h>
#include <hip/hip_bf16.h>

typedef _Float16 f16;
typedef __attribute__((ext_vector_type(16))) _Float16 v16h;
typedef __attribute__((ext_vector_type(8)))  _Float16 v8h;
typedef __attribute__((ext_vector_type(8)))  float    v8f;
typedef __attribute__((ext_vector_type(4)))  float    v4f;
typedef __attribute__((ext_vector_type(4)))  int      v4i;

#define DIMN     256
#define S_TOTAL  4096   // per batch (64*64)
#define NBATCH   2
#define NHEADS   8
#define DHEAD    32
#define ROWS_ALL 8192   // NBATCH * S_TOTAL
#define PLD      40     // padded LDS row stride (halfs): 80B, 16B-aligned

// ---- CDNA5 async global->LDS path (guarded: falls back to load+ds_store) --
#if defined(__has_builtin)
# if __has_builtin(__builtin_amdgcn_global_load_async_to_lds_b128)
#  define HAVE_ASYNC_LDS 1
# endif
# if __has_builtin(__builtin_amdgcn_s_wait_asynccnt)
#  define HAVE_WAIT_ASYNC 1
# endif
#endif
#ifndef HAVE_ASYNC_LDS
# define HAVE_ASYNC_LDS 0
#endif
#ifndef HAVE_WAIT_ASYNC
# define HAVE_WAIT_ASYNC 0
#endif

__device__ __forceinline__ void lds_copy_b128(const f16* __restrict__ g, f16* l) {
#if HAVE_ASYNC_LDS
  __builtin_amdgcn_global_load_async_to_lds_b128(
      (v4i*)g, (v4i*)l, /*offset=*/0, /*cpol=*/0);
#else
  *(v8h*)l = *(const v8h*)g;
#endif
}

__device__ __forceinline__ void wait_async0() {
#if HAVE_ASYNC_LDS
# if HAVE_WAIT_ASYNC
  __builtin_amdgcn_s_wait_asynccnt(0);
# else
  asm volatile("s_wait_asynccnt 0x0" ::: "memory");
# endif
#endif
}

// ---------------- WMMA + fragment helpers (CDNA5 16x16x32 f16) -------------

__device__ __forceinline__ v8f wmma_f16(v16h a, v16h b, v8f c) {
  return __builtin_amdgcn_wmma_f32_16x16x32_f16(
      /*neg_a=*/false, a, /*neg_b=*/false, b,
      /*c_mod=*/(short)0, c, /*reuse_a=*/false, /*reuse_b=*/false);
}

// A-matrix 16x32 (MxK), row-major, leading dim `ld` (halfs).
// Lane l: M=l&15; per-lane halfs are two contiguous 16B runs: K=8*hh+[0..7], 16+8*hh+[0..7].
__device__ __forceinline__ v16h frag_a_f16(const f16* __restrict__ p, int ld, int lane) {
  int m = lane & 15, hh = (lane >> 4) & 1;
  const f16* base = p + (size_t)m * ld + 8 * hh;
  v8h lo = *(const v8h*)base;
  v8h hi = *(const v8h*)(base + 16);
  return __builtin_shufflevector(lo, hi, 0,1,2,3,4,5,6,7,8,9,10,11,12,13,14,15);
}

// Same from f32 source (two 32-byte runs + convert).
__device__ __forceinline__ v16h frag_a_f32(const float* __restrict__ p, int ld, int lane) {
  int m = lane & 15, hh = (lane >> 4) & 1;
  const float* base = p + (size_t)m * ld + 8 * hh;
  v4f a0 = *(const v4f*)base;
  v4f a1 = *(const v4f*)(base + 4);
  v4f b0 = *(const v4f*)(base + 16);
  v4f b1 = *(const v4f*)(base + 20);
  v16h a;
#pragma unroll
  for (int i = 0; i < 4; ++i) {
    a[i]      = (f16)a0[i];
    a[4 + i]  = (f16)a1[i];
    a[8 + i]  = (f16)b0[i];
    a[12 + i] = (f16)b1[i];
  }
  return a;
}

// B-matrix 32x16 (KxN), element (k,n) at p[k + n*sn] (K contiguous per column).
// Lane l: N=l&15, K=8*hh+[0..7], 16+8*hh+[0..7] : two contiguous 16-byte runs per lane.
__device__ __forceinline__ v16h frag_b_kc(const f16* __restrict__ p, size_t sn, int lane) {
  int n = lane & 15, hh = (lane >> 4) & 1;
  const f16* base = p + (size_t)n * sn + 8 * hh;
  v8h lo = *(const v8h*)base;
  v8h hi = *(const v8h*)(base + 16);
  return __builtin_shufflevector(lo, hi, 0,1,2,3,4,5,6,7,8,9,10,11,12,13,14,15);
}

// ---------------- Kernel 1: f32 -> f16 transposed weight conversion --------

__global__ void cvt_transpose(const float* __restrict__ src, f16* __restrict__ dst,
                              int K, int N) {
  int i = blockIdx.x * blockDim.x + threadIdx.x;
  if (i < K * N) {
    int k = i / N, n = i % N;
    dst[(size_t)n * K + k] = (f16)src[i];
  }
}

// ---------------- Kernel 2: fused QKV projection ---------------------------
// q = (query @ Wq + bq) * dh^-0.5 ; [k|v] = sim @ Wkv + bkv ; V stored transposed.

__global__ void __launch_bounds__(256) qkv_kernel(
    const float* __restrict__ query, const float* __restrict__ sim,
    const f16* __restrict__ wqT, const f16* __restrict__ wkvT,
    const float* __restrict__ bq, const float* __restrict__ bkv,
    f16* __restrict__ qb, f16* __restrict__ kbuf, f16* __restrict__ vt) {
  int lane = threadIdx.x & 31;
  int wid  = threadIdx.x >> 5;
  int t  = blockIdx.x * 8 + wid;     // 0 .. 24575
  int rt = t / 48, ct = t % 48;
  int row0 = rt * 16;

  const float* act; const f16* wT; const float* bias;
  int wcol; float scale; bool isV = false; f16* dst = nullptr; int dcol = 0;
  if (ct < 16) {
    act = query; wT = wqT; bias = bq;
    wcol = ct * 16; dst = qb; dcol = wcol;
    scale = 0.17677669529663688f;    // dh^-0.5 folded into q
  } else {
    act = sim; wT = wkvT; bias = bkv;
    wcol = (ct - 16) * 16; scale = 1.0f;
    if (wcol < DIMN) { dst = kbuf; dcol = wcol; }
    else             { isV = true;  dcol = wcol - DIMN; }
  }

  v8f c = {};
#pragma unroll
  for (int kc = 0; kc < DIMN; kc += 32) {
    v16h a = frag_a_f32(act + (size_t)row0 * DIMN + kc, DIMN, lane);
    v16h b = frag_b_kc(wT + (size_t)wcol * DIMN + kc, DIMN, lane);
    c = wmma_f16(a, b, c);
  }

  int n = lane & 15, hh = (lane >> 4) & 1;
  float bv = bias[wcol + n];
  if (!isV) {
#pragma unroll
    for (int r = 0; r < 8; ++r) {
      float val = (c[r] + bv) * scale;
      dst[(size_t)(row0 + r + 8 * hh) * DIMN + dcol + n] = (f16)val;
    }
  } else {
    int bbi = row0 >> 12, key0 = row0 & (S_TOTAL - 1);
    v8h pack;
#pragma unroll
    for (int r = 0; r < 8; ++r) pack[r] = (f16)(c[r] + bv);
    f16* vcol = vt + ((size_t)bbi * DIMN + dcol + n) * S_TOTAL + key0 + 8 * hh;
    *(v8h*)vcol = pack;              // contiguous 16B store per lane
  }
}

// ---------------- Kernel 3: flash attention --------------------------------
// Block = 8 waves sharing one (batch, head); each wave owns one 16-query tile.
// K/V chunks (32 keys) are double-buffered in LDS via async global->LDS loads.

__global__ void __launch_bounds__(256) attn_kernel(
    const f16* __restrict__ qb, const f16* __restrict__ kbuf,
    const f16* __restrict__ vt, f16* __restrict__ xb) {
  __shared__ f16 Ks[2][32][32];      // [buf][key][dh]
  __shared__ f16 Vs[2][32][32];      // [buf][dh][key]
  __shared__ f16 P[8][16][PLD];      // per-wave probability tile

  int tid  = threadIdx.x;
  int lane = tid & 31;
  int wid  = tid >> 5;
  int bt   = blockIdx.x;             // 0 .. 511
  int bbi  = bt >> 8;                // batch   (256 blocks per batch)
  int h    = (bt >> 5) & 7;          // head    (32 blocks per head)
  int qt   = (bt & 31) * 8 + wid;    // query tile 0..255
  int row0 = qt * 16;

  const f16* qp  = qb   + ((size_t)bbi * S_TOTAL + row0) * DIMN + h * DHEAD;
  const f16* kcp = kbuf + (size_t)bbi * S_TOTAL * DIMN + h * DHEAD;         // [key][dh]
  const f16* vcp = vt   + ((size_t)bbi * DIMN + h * DHEAD) * S_TOTAL;       // [dh][key]

  // Cooperative stage of one 32-key K+V chunk: 256 threads x 16B = 4KB.
  // Waves 0-3 stage K, waves 4-7 stage V (wave-uniform split).
  int sHalf = tid >> 7;              // 0 = K, 1 = V
  int sIdx  = (tid & 127) >> 2;      // key (K) or dh (V)
  int sSeg  = (tid & 3) * 8;         // 8-half segment

  v16h aq = frag_a_f16(qp, DIMN, lane);     // 16 queries x 32(dh)

  // Prologue: issue chunk 0 into buffer 0.
  {
    const f16* src = (sHalf == 0) ? (kcp + (size_t)sIdx * DIMN + sSeg)
                                  : (vcp + (size_t)sIdx * S_TOTAL + sSeg);
    f16* dstl = (sHalf == 0) ? &Ks[0][sIdx][sSeg] : &Vs[0][sIdx][sSeg];
    lds_copy_b128(src, dstl);
  }

  v8f acc0 = {}, acc1 = {};
  float mrow[8], lsum[8];
#pragma unroll
  for (int r = 0; r < 8; ++r) { mrow[r] = -1e30f; lsum[r] = 0.f; }

  int n = lane & 15, hh = (lane >> 4) & 1;

  for (int kc = 0; kc < S_TOTAL; kc += 32) {
    int cur = (kc >> 5) & 1;
    wait_async0();                   // current chunk's LDS writes landed
    __syncthreads();                 // visible to all waves; prev buffer free

    if (kc + 32 < S_TOTAL) {         // issue next chunk into other buffer
      int nb = 1 - cur;
      const f16* src = (sHalf == 0) ? (kcp + (size_t)(kc + 32 + sIdx) * DIMN + sSeg)
                                    : (vcp + (size_t)sIdx * S_TOTAL + kc + 32 + sSeg);
      f16* dstl = (sHalf == 0) ? &Ks[nb][sIdx][sSeg] : &Vs[nb][sIdx][sSeg];
      lds_copy_b128(src, dstl);
    }

    // scores: Q(16x32) @ K^T(32x16) x2 — B operands K(dh)-contiguous in LDS
    v16h bk0 = frag_b_kc(&Ks[cur][0][0],  32, lane);   // keys kc..+15
    v16h bk1 = frag_b_kc(&Ks[cur][16][0], 32, lane);   // keys +16..+31
    v8f z = {};
    v8f s0 = wmma_f16(aq, bk0, z);
    v8f s1 = wmma_f16(aq, bk1, z);

    // online softmax; C layout: lane = column (within half), reg r = row r+8*hh
#pragma unroll
    for (int r = 0; r < 8; ++r) {
      float cm = fmaxf(s0[r], s1[r]);
#pragma unroll
      for (int off = 1; off < 16; off <<= 1)
        cm = fmaxf(cm, __shfl_xor(cm, off, 32));
      float mn    = fmaxf(mrow[r], cm);
      float alpha = __expf(mrow[r] - mn);
      float p0 = __expf(s0[r] - mn);
      float p1 = __expf(s1[r] - mn);
      float ps = p0 + p1;
#pragma unroll
      for (int off = 1; off < 16; off <<= 1)
        ps += __shfl_xor(ps, off, 32);
      lsum[r] = lsum[r] * alpha + ps;
      mrow[r] = mn;
      acc0[r] *= alpha;
      acc1[r] *= alpha;
      P[wid][r + 8 * hh][n]      = (f16)p0;
      P[wid][r + 8 * hh][n + 16] = (f16)p1;
    }
    __syncthreads();
    v16h ap  = frag_a_f16(&P[wid][0][0], PLD, lane);   // 2x ds_load_b128
    v16h bv0 = frag_b_kc(&Vs[cur][0][0],  32, lane);   // dh 0..15
    v16h bv1 = frag_b_kc(&Vs[cur][16][0], 32, lane);   // dh 16..31
    acc0 = wmma_f16(ap, bv0, acc0);
    acc1 = wmma_f16(ap, bv1, acc1);
  }

  f16* xp = xb + ((size_t)bbi * S_TOTAL + row0) * DIMN + h * DHEAD;
#pragma unroll
  for (int r = 0; r < 8; ++r) {
    float inv = 1.0f / lsum[r];
    xp[(size_t)(r + 8 * hh) * DIMN + n]      = (f16)(acc0[r] * inv);
    xp[(size_t)(r + 8 * hh) * DIMN + n + 16] = (f16)(acc1[r] * inv);
  }
}

// ---------------- Kernel 4: output projection ------------------------------

__global__ void __launch_bounds__(256) proj_kernel(
    const f16* __restrict__ xb, const f16* __restrict__ wpT,
    const float* __restrict__ bp, float* __restrict__ out) {
  int lane = threadIdx.x & 31;
  int wid  = threadIdx.x >> 5;
  int t  = blockIdx.x * 8 + wid;     // 0 .. 8191
  int rt = t >> 4, ct = t & 15;
  int row0 = rt * 16, col0 = ct * 16;

  v8f c = {};
#pragma unroll
  for (int kc = 0; kc < DIMN; kc += 32) {
    v16h a = frag_a_f16(xb + (size_t)row0 * DIMN + kc, DIMN, lane);
    v16h b = frag_b_kc(wpT + (size_t)col0 * DIMN + kc, DIMN, lane);
    c = wmma_f16(a, b, c);
  }

  int n = lane & 15, hh = (lane >> 4) & 1;
  float bv = bp[col0 + n];
#pragma unroll
  for (int r = 0; r < 8; ++r)
    out[(size_t)(row0 + r + 8 * hh) * DIMN + col0 + n] = c[r] + bv;
}

// ---------------- Launcher -------------------------------------------------

extern "C" void kernel_launch(void* const* d_in, const int* in_sizes, int n_in,
                              void* d_out, int out_size, void* d_ws, size_t ws_size,
                              hipStream_t stream) {
  const float* query = (const float*)d_in[0];
  const float* sim   = (const float*)d_in[1];
  const float* Wq    = (const float*)d_in[2];
  const float* bq    = (const float*)d_in[3];
  const float* Wkv   = (const float*)d_in[4];
  const float* bkv   = (const float*)d_in[5];
  const float* Wp    = (const float*)d_in[6];
  const float* bp    = (const float*)d_in[7];
  float* out = (float*)d_out;

  char* ws = (char*)d_ws;
  f16* wqT  = (f16*)ws; ws += (size_t)DIMN * DIMN * 2;
  f16* wkvT = (f16*)ws; ws += (size_t)DIMN * 2 * DIMN * 2;
  f16* wpT  = (f16*)ws; ws += (size_t)DIMN * DIMN * 2;
  f16* qb   = (f16*)ws; ws += (size_t)ROWS_ALL * DIMN * 2;
  f16* kbuf = (f16*)ws; ws += (size_t)ROWS_ALL * DIMN * 2;
  f16* vt   = (f16*)ws; ws += (size_t)ROWS_ALL * DIMN * 2;
  f16* xb   = (f16*)ws; ws += (size_t)ROWS_ALL * DIMN * 2;

  cvt_transpose<<<(DIMN * DIMN + 255) / 256, 256, 0, stream>>>(Wq, wqT, DIMN, DIMN);
  cvt_transpose<<<(DIMN * 2 * DIMN + 255) / 256, 256, 0, stream>>>(Wkv, wkvT, DIMN, 2 * DIMN);
  cvt_transpose<<<(DIMN * DIMN + 255) / 256, 256, 0, stream>>>(Wp, wpT, DIMN, DIMN);

  qkv_kernel<<<3072, 256, 0, stream>>>(query, sim, wqT, wkvT, bq, bkv, qb, kbuf, vt);
  attn_kernel<<<512, 256, 0, stream>>>(qb, kbuf, vt, xb);
  proj_kernel<<<1024, 256, 0, stream>>>(xb, wpT, bp, out);
}
